// METST_SF_33852932227257
// MI455X (gfx1250) — matrix operand order repaired and checked by reading, not compiled, not run on hardware
//
#include <hip/hip_runtime.h>
#include <hip/hip_bf16.h>
#include <math.h>

typedef __attribute__((ext_vector_type(16))) _Float16 v16h;
typedef __attribute__((ext_vector_type(8)))  _Float16 h8;
typedef __attribute__((ext_vector_type(8)))  float    v8f;

union V16H { v16h v; h8 h[2]; _Float16 e[16]; };
union V8F  { v8f  v; float e[8]; };

static inline int ceildiv(int a, int b) { return (a + b - 1) / b; }
static inline int roundup32(int a) { return (a + 31) & ~31; }

__device__ __forceinline__ float sigmoidf_(float x) { return 1.f / (1.f + __expf(-x)); }

// ---------------- pack A: f32[M,srcStride-strided rows, K cols] -> f16[M,Kp], zero pad ----------------
__global__ void k_pack_a(const float* __restrict__ src, int srcStride,
                         _Float16* __restrict__ dst, int M, int K, int Kp) {
    int idx = blockIdx.x * blockDim.x + threadIdx.x;
    if (idx >= M * Kp) return;
    int m = idx / Kp, k = idx % Kp;
    dst[idx] = (k < K) ? (_Float16)src[(long)m * srcStride + k] : (_Float16)0.f;
}

// ---------------- pack B transposed: f32 B[K,N] -> f16 BT[Np,Kp], zero pad ----------------
__global__ void k_pack_bt(const float* __restrict__ src,
                          _Float16* __restrict__ dst, int K, int N, int Kp, int Np) {
    int idx = blockIdx.x * blockDim.x + threadIdx.x;
    if (idx >= Np * Kp) return;
    int n = idx / Kp, k = idx % Kp;
    dst[idx] = (k < K && n < N) ? (_Float16)src[(long)k * N + n] : (_Float16)0.f;
}

// ---------------- WMMA GEMM: C[M,N] = A[M,K] @ B[K,N] ----------------
// A packed [M,Kp]; B packed transposed [Np,Kp]; Kp,Np multiples of 32; M multiple of 16.
// One wave per 16x32 output tile: 2 wmma sharing the A fragment. No guards in the K loop.
__global__ void k_gemm_wmma(const _Float16* __restrict__ A,
                            const _Float16* __restrict__ BT,
                            float* __restrict__ C, int ldc,
                            int M, int N, int Kp) {
    int lane = threadIdx.x & 31;
    int col  = lane & 15;
    int hi   = lane >> 4;
    int n0 = blockIdx.x * 32;
    int m0 = blockIdx.y * 16;

    const _Float16* __restrict__ arow  = A  + (long)(m0 + col) * Kp;
    const _Float16* __restrict__ brow0 = BT + (long)(n0 + col) * Kp;
    const _Float16* __restrict__ brow1 = BT + (long)(n0 + 16 + col) * Kp;

    v8f acc0 = {}, acc1 = {};
    for (int k0 = 0; k0 < Kp; k0 += 32) {
        V16H ua, ub0, ub1;
        // A fragment: A row m, halves [k0+hi*16, +16) (same k order as B)
        ua.h[0]  = *(const h8*)(arow + k0 + hi * 16);
        ua.h[1]  = *(const h8*)(arow + k0 + hi * 16 + 8);
        // B fragment: BT row n, halves [k0+hi*16, +16)
        ub0.h[0] = *(const h8*)(brow0 + k0 + hi * 16);
        ub0.h[1] = *(const h8*)(brow0 + k0 + hi * 16 + 8);
        ub1.h[0] = *(const h8*)(brow1 + k0 + hi * 16);
        ub1.h[1] = *(const h8*)(brow1 + k0 + hi * 16 + 8);
        acc0 = __builtin_amdgcn_wmma_f32_16x16x32_f16(false, ua.v, false, ub0.v,
                                                      (short)0, acc0, false, false);
        acc1 = __builtin_amdgcn_wmma_f32_16x16x32_f16(false, ua.v, false, ub1.v,
                                                      (short)0, acc1, false, false);
    }
    V8F uc0, uc1; uc0.v = acc0; uc1.v = acc1;
    int n_a = n0 + col, n_b = n0 + 16 + col;
#pragma unroll
    for (int r0 = 0; r0 < 8; ++r0) {
        int m = m0 + r0 + hi * 8;
        if (n_a < N) C[(long)m * ldc + n_a] = uc0.e[r0];
        if (n_b < N) C[(long)m * ldc + n_b] = uc1.e[r0];
    }
}

// ---------------- Conv1d (k=3, pad=1) + BN + ReLU, layout flags ----------------
__global__ void k_conv_bn_relu(const float* __restrict__ in, const float* __restrict__ w,
                               const float* __restrict__ cb, const float* __restrict__ g,
                               const float* __restrict__ bb, const float* __restrict__ mm,
                               const float* __restrict__ vv, float* __restrict__ out,
                               int Cin, int Cout, int Lin, int Lout, int stride,
                               int in_trans, int out_trans, int accum) {
    int idx = blockIdx.x * blockDim.x + threadIdx.x;
    if (idx >= Cout * Lout) return;
    int co = idx / Lout, l = idx % Lout;
    float s = cb[co];
    for (int ci = 0; ci < Cin; ++ci) {
        const float* wr = &w[((long)co * Cin + ci) * 3];
        for (int k = 0; k < 3; ++k) {
            int li = l * stride + k - 1;
            if (li >= 0 && li < Lin) {
                float xv = in_trans ? in[(long)li * Cin + ci] : in[(long)ci * Lin + li];
                s += wr[k] * xv;
            }
        }
    }
    s = (s - mm[co]) * g[co] * rsqrtf(vv[co] + 1e-5f) + bb[co];
    s = fmaxf(s, 0.f);
    if (out_trans) { if (accum) out[(long)l * Cout + co] += s; else out[(long)l * Cout + co] = s; }
    else out[(long)co * Lout + l] = s;
}

__global__ void k_transpose(const float* __restrict__ in, float* __restrict__ out, int C, int L) {
    int idx = blockIdx.x * blockDim.x + threadIdx.x;
    if (idx >= C * L) return;
    int c = idx / L, l = idx % L;
    out[(long)l * C + c] = in[(long)c * L + l];
}

// ---------------- LayerNorm over last dim (block per row, blockDim == C) ----------------
__global__ void k_layernorm(const float* __restrict__ x, const float* __restrict__ w,
                            const float* __restrict__ b, float* __restrict__ o, int C) {
    int row = blockIdx.x, t = threadIdx.x;
    __shared__ float s1[256], s2[256];
    float v = x[(long)row * C + t];
    s1[t] = v; s2[t] = v * v;
    __syncthreads();
    for (int s = C >> 1; s > 0; s >>= 1) {
        if (t < s) { s1[t] += s1[t + s]; s2[t] += s2[t + s]; }
        __syncthreads();
    }
    float m = s1[0] / C;
    float var = s2[0] / C - m * m;
    float inv = rsqrtf(var + 1e-5f);
    o[(long)row * C + t] = (v - m) * inv * w[t] + b[t];
}

// x[row] += LN(hin[row])
__global__ void k_ln_residual(const float* __restrict__ hin, const float* __restrict__ w,
                              const float* __restrict__ b, float* __restrict__ x, int C) {
    int row = blockIdx.x, t = threadIdx.x;
    __shared__ float s1[256], s2[256];
    float v = hin[(long)row * C + t];
    s1[t] = v; s2[t] = v * v;
    __syncthreads();
    for (int s = C >> 1; s > 0; s >>= 1) {
        if (t < s) { s1[t] += s1[t + s]; s2[t] += s2[t + s]; }
        __syncthreads();
    }
    float m = s1[0] / C;
    float var = s2[0] / C - m * m;
    float inv = rsqrtf(var + 1e-5f);
    x[(long)row * C + t] += (v - m) * inv * w[t] + b[t];
}

// ---------------- Depthwise causal conv (K=4) + SiLU; reads xc-half of xz ----------------
__global__ void k_dwconv_silu(const float* __restrict__ xz, const float* __restrict__ cw,
                              const float* __restrict__ cb, float* __restrict__ xc,
                              int L, int di) {
    int idx = blockIdx.x * blockDim.x + threadIdx.x;
    if (idx >= L * di) return;
    int l = idx / di, c = idx % di;
    float s = cb[c];
#pragma unroll
    for (int k = 0; k < 4; ++k) {
        int li = l + k - 3;
        if (li >= 0) s += cw[c * 4 + k] * xz[(long)li * (2 * di) + c];
    }
    xc[idx] = s * sigmoidf_(s);
}

__global__ void k_softplus_bias(const float* __restrict__ din, const float* __restrict__ dtb,
                                float* __restrict__ dlt, int L, int di) {
    int idx = blockIdx.x * blockDim.x + threadIdx.x;
    if (idx >= L * di) return;
    int c = idx % di;
    float x = din[idx] + dtb[c];
    dlt[idx] = (x > 20.f) ? x : log1pf(__expf(x));
}

// ---------------- Selective scan: block per channel, thread per state (N=64) ----------------
__global__ void k_scan(const float* __restrict__ delta, const float* __restrict__ xc,
                       const float* __restrict__ dbl, int r, int S,
                       const float* __restrict__ A_log, const float* __restrict__ Dp,
                       const float* __restrict__ xz, float* __restrict__ y,
                       int L, int di) {
    int c = blockIdx.x;
    int n = threadIdx.x;  // 0..63
    __shared__ float red[64];
    float A  = -__expf(A_log[(long)c * 64 + n]);
    float Dc = Dp[c];
    float h = 0.f;
    for (int l = 0; l < L; ++l) {
        float dlt = delta[(long)l * di + c];
        float xcv = xc[(long)l * di + c];
        float bm  = dbl[(long)l * S + r + n];
        float cm  = dbl[(long)l * S + r + 64 + n];
        h = __expf(dlt * A) * h + dlt * xcv * bm;
        red[n] = h * cm;
        __syncthreads();
        for (int s = 32; s > 0; s >>= 1) {
            if (n < s) red[n] += red[n + s];
            __syncthreads();
        }
        if (n == 0) {
            float yy = red[0] + xcv * Dc;
            float zz = xz[(long)l * 2 * di + di + c];
            y[(long)l * di + c] = yy * zz * sigmoidf_(zz);
        }
        __syncthreads();
    }
}

// ---------------- Heads ----------------
__global__ void k_heads(const float* __restrict__ xs, const float* __restrict__ xf,
                        const float* __restrict__ sw, const float* __restrict__ sb,
                        const float* __restrict__ rw, const float* __restrict__ rb,
                        float* __restrict__ out) {
    int idx = blockIdx.x * blockDim.x + threadIdx.x;
    if (idx >= 512 * 4) return;
    int l = idx >> 2, j = idx & 3;
    const float* xsr = &xs[(long)(l >> 1) * 256];
    const float* xfr = &xf[(long)l * 128];
    float s = 0.f;
    if (j == 0) {
        for (int ci = 0; ci < 256; ++ci) s += xsr[ci] * sw[ci];
        for (int ci = 0; ci < 128; ++ci) s += xfr[ci] * sw[256 + ci];
        out[l] = sigmoidf_(s + sb[0]);
    } else {
        int jj = j - 1;
        for (int ci = 0; ci < 256; ++ci) s += xsr[ci] * rw[ci * 3 + jj];
        for (int ci = 0; ci < 128; ++ci) s += xfr[ci] * rw[(256 + ci) * 3 + jj];
        out[512 + l * 3 + jj] = s + rb[jj];
    }
}

// ================= host side =================
struct WS {
    float *xs, *xf, *h0, *h1, *ln, *xz, *xc, *dbl, *dproj, *delta, *y, *mout;
    _Float16 *a16, *b16;
};
struct BlockP {
    const float *lnm_w, *lnm_b, *ln1_w, *ln1_b, *in_w, *conv_w, *conv_b,
                *xproj_w, *dt_w, *dt_b, *A_log, *D, *out_w;
};

static BlockP get_block(void* const* d_in, int i) {
    int b = 19 + i * 13;
    BlockP p;
    p.lnm_w   = (const float*)d_in[b + 0];
    p.lnm_b   = (const float*)d_in[b + 1];
    p.ln1_w   = (const float*)d_in[b + 2];
    p.ln1_b   = (const float*)d_in[b + 3];
    p.in_w    = (const float*)d_in[b + 4];
    p.conv_w  = (const float*)d_in[b + 5];
    p.conv_b  = (const float*)d_in[b + 6];
    p.xproj_w = (const float*)d_in[b + 7];
    p.dt_w    = (const float*)d_in[b + 8];
    p.dt_b    = (const float*)d_in[b + 9];
    p.A_log   = (const float*)d_in[b + 10];
    p.D       = (const float*)d_in[b + 11];
    p.out_w   = (const float*)d_in[b + 12];
    return p;
}

// C = A(f32,[M,Ksrc-strided]) @ B(f32,[K,N]) via f16 WMMA, staged+padded in workspace
static void gemm(hipStream_t st, const WS& w,
                 const float* A, int srcStride, const float* B,
                 float* C, int ldc, int M, int N, int K) {
    int Kp = roundup32(K);
    int Np = roundup32(N);
    k_pack_a<<<ceildiv(M * Kp, 256), 256, 0, st>>>(A, srcStride, w.a16, M, K, Kp);
    k_pack_bt<<<ceildiv(Np * Kp, 256), 256, 0, st>>>(B, w.b16, K, N, Kp, Np);
    k_gemm_wmma<<<dim3(ceildiv(N, 32), M / 16), 32, 0, st>>>(w.a16, w.b16, C, ldc, M, N, Kp);
}

static void run_block(hipStream_t st, float* x, int L, int d, const BlockP& p, const WS& w) {
    const int di = 2 * d, r = (d + 15) / 16, Nd = r + 128, S = 144;
    // LN
    k_layernorm<<<L, d, 0, st>>>(x, p.lnm_w, p.lnm_b, w.ln, d);
    // in_proj: xz = LN(x) @ in_w   (L x 2di)
    gemm(st, w, w.ln, d, p.in_w, w.xz, 2 * di, L, 2 * di, d);
    // depthwise causal conv + silu -> xc
    k_dwconv_silu<<<ceildiv(L * di, 256), 256, 0, st>>>(w.xz, p.conv_w, p.conv_b, w.xc, L, di);
    // xproj: dbl = xc @ xproj_w   (L x Nd), row stride S
    gemm(st, w, w.xc, di, p.xproj_w, w.dbl, S, L, Nd, di);
    // dt projection: dproj = dbl[:, :r] @ dt_w   (L x di); A is strided slice of dbl
    gemm(st, w, w.dbl, S, p.dt_w, w.dproj, di, L, di, r);
    k_softplus_bias<<<ceildiv(L * di, 256), 256, 0, st>>>(w.dproj, p.dt_b, w.delta, L, di);
    // scan + D-term + silu(z) gate
    k_scan<<<di, 64, 0, st>>>(w.delta, w.xc, w.dbl, r, S, p.A_log, p.D, w.xz, w.y, L, di);
    // out_proj + residual LN
    gemm(st, w, w.y, di, p.out_w, w.mout, d, L, d, di);
    k_ln_residual<<<L, d, 0, st>>>(w.mout, p.ln1_w, p.ln1_b, x, d);
}

extern "C" void kernel_launch(void* const* d_in, const int* in_sizes, int n_in,
                              void* d_out, int out_size, void* d_ws, size_t ws_size,
                              hipStream_t stream) {
    (void)in_sizes; (void)n_in; (void)out_size; (void)ws_size;
    const float* X = (const float*)d_in[0];  // (36, 512)
    auto F = [&](int i) { return (const float*)d_in[i]; };

    float* base = (float*)d_ws;
    size_t off = 0;
    auto alloc = [&](size_t n) { float* p = base + off; off += n; return p; };
    WS w;
    w.xs = alloc(65536);  w.xf = alloc(65536);  w.h0 = alloc(65536);  w.h1 = alloc(65536);
    w.ln = alloc(65536);  w.xz = alloc(262144); w.xc = alloc(131072); w.dbl = alloc(73728);
    w.dproj = alloc(131072); w.delta = alloc(131072); w.y = alloc(131072); w.mout = alloc(65536);
    w.a16 = (_Float16*)(base + off); off += 131072 / 2;   // max M*Kp = 256*512 / 512*256
    w.b16 = (_Float16*)(base + off); off += 262144 / 2;   // max Np*Kp = 1024*256

    // stem1: (36,512) -> h0 (128,512), ch-major
    k_conv_bn_relu<<<ceildiv(128 * 512, 256), 256, 0, stream>>>(
        X, F(1), F(2), F(3), F(4), F(5), F(6), w.h0, 36, 128, 512, 512, 1, 0, 0, 0);
    // stem2: h0 -> h1 (128,512)
    k_conv_bn_relu<<<ceildiv(128 * 512, 256), 256, 0, stream>>>(
        w.h0, F(7), F(8), F(9), F(10), F(11), F(12), w.h1, 128, 128, 512, 512, 1, 0, 0, 0);
    // xf = h1^T : (512,128)
    k_transpose<<<ceildiv(128 * 512, 256), 256, 0, stream>>>(w.h1, w.xf, 128, 512);
    // slow conv (stride 2) -> xs (256,256) L-major
    k_conv_bn_relu<<<ceildiv(256 * 256, 256), 256, 0, stream>>>(
        w.h1, F(13), F(14), F(15), F(16), F(17), F(18), w.xs, 128, 256, 512, 256, 2, 0, 1, 0);

    // stage 1
    for (int i = 0; i < 4; ++i)  { BlockP p = get_block(d_in, i);      run_block(stream, w.xs, 256, 256, p, w); }
    for (int i = 12; i < 16; ++i){ BlockP p = get_block(d_in, i);      run_block(stream, w.xf, 512, 128, p, w); }
    // fuse1: xs += relu(bn(conv_s2(xf^T)))
    k_conv_bn_relu<<<ceildiv(256 * 256, 256), 256, 0, stream>>>(
        w.xf, F(331), F(332), F(333), F(334), F(335), F(336), w.xs, 128, 256, 512, 256, 2, 1, 1, 1);

    // stage 2
    for (int i = 4; i < 8; ++i)  { BlockP p = get_block(d_in, i);      run_block(stream, w.xs, 256, 256, p, w); }
    for (int i = 16; i < 20; ++i){ BlockP p = get_block(d_in, i);      run_block(stream, w.xf, 512, 128, p, w); }
    k_conv_bn_relu<<<ceildiv(256 * 256, 256), 256, 0, stream>>>(
        w.xf, F(337), F(338), F(339), F(340), F(341), F(342), w.xs, 128, 256, 512, 256, 2, 1, 1, 1);

    // stage 3
    for (int i = 8; i < 12; ++i) { BlockP p = get_block(d_in, i);      run_block(stream, w.xs, 256, 256, p, w); }
    for (int i = 20; i < 24; ++i){ BlockP p = get_block(d_in, i);      run_block(stream, w.xf, 512, 128, p, w); }

    // heads -> d_out (spot[512], recog[512*3])
    k_heads<<<ceildiv(512 * 4, 256), 256, 0, stream>>>(
        w.xs, w.xf, F(343), F(344), F(345), F(346), (float*)d_out);
}
